// CrossAttentionFusion_9216999817754
// MI455X (gfx1250) — matrix operand order repaired and checked by reading, not compiled, not run on hardware
//
#include <hip/hip_runtime.h>

// ---------------------------------------------------------------------------
// CrossAttentionFusion on gfx1250 (MI455X):
//   phase 0: one-shot f32 -> bf16 conversion of Wq/Wk/Wv (L2-resident)
//   phase 1: Q/K/V 1x1-conv projections -> bf16 (WMMA f32_16x16x32_bf16)
//   phase 2: flash attention (online softmax) + gamma*out + residual
// All matmuls use v_wmma_f32_16x16x32_bf16 (wave32, 16x16x32).
// ---------------------------------------------------------------------------

#define BATCH 4
#define CHN   256
#define HWTOK 4096

typedef __attribute__((ext_vector_type(16))) __bf16 v16bf;
typedef __attribute__((ext_vector_type(8)))  float  v8f;

union Frag {
    v16bf           v;
    uint4           q[2];
    unsigned short  u[16];
};

// f32 -> bf16 via the native conversion (clang lowers fptrunc-to-bf16 to the
// hardware v_cvt when available on gfx1250; RNE either way).
static __device__ __forceinline__ unsigned short f2bf(float f) {
    __bf16 h = (__bf16)f;
    return __builtin_bit_cast(unsigned short, h);
}

static __device__ __forceinline__ v8f wmma_bf16(v16bf a, v16bf b, v8f c) {
    // (neg_a, A, neg_b, B, c_mod, C, reuse_a, reuse_b)
    return __builtin_amdgcn_wmma_f32_16x16x32_bf16(false, a, false, b,
                                                   (short)0, c, false, false);
}

// A fragment (16x32 bf16): lane -> row = lane%16, g = lane/16 selects the
// 16 contiguous K values [g*16, g*16+16)
static __device__ __forceinline__ v16bf loadA_bf(const unsigned short* row, int g) {
    Frag f;
    f.q[0] = *reinterpret_cast<const uint4*>(row + g * 16);
    f.q[1] = *reinterpret_cast<const uint4*>(row + g * 16 + 8);
    return f.v;
}

// B fragment (32x16 bf16): lane -> col = lane%16, per-lane 16 contiguous K
// values at K = 16*g .. 16*g+15. `p` already points at (col row) + 16*g.
static __device__ __forceinline__ v16bf loadB_bf(const unsigned short* p) {
    Frag f;
    f.q[0] = *reinterpret_cast<const uint4*>(p);
    f.q[1] = *reinterpret_cast<const uint4*>(p + 8);
    return f.v;
}

// ---------------------------------------------------------------------------
// Phase 0: convert the three 256x256 f32 weight matrices to bf16 (row-major).
// ---------------------------------------------------------------------------
__global__ __launch_bounds__(256)
void prep_weights(const float* __restrict__ Wq, const float* __restrict__ Wk,
                  const float* __restrict__ Wv, unsigned short* __restrict__ Wb) {
    const int i = blockIdx.x * blockDim.x + threadIdx.x;   // 0 .. 3*65536-1
    const int which = i >> 16;                             // 0..2
    const int idx   = i & 0xffff;
    const float* W  = (which == 0 ? Wq : (which == 1 ? Wk : Wv));
    Wb[i] = f2bf(W[idx]);
}

// ---------------------------------------------------------------------------
// Phase 1: projections.
//   mat 0: Qt[b][n][o] = sum_c Wq[o][c]*pre[b][c][n] + bq[o]   (token-major)
//   mat 1: Kt[b][n][o] = ... post/Wk/bk                        (token-major)
//   mat 2: Vc[b][o][n] = ... post/Wv/bv                        (channel-major)
// Block: 256 thr (8 waves) computes 64 tokens x 256 out-channels.
// ---------------------------------------------------------------------------
__global__ __launch_bounds__(256)
void proj_bf16_wmma(const float* __restrict__ pre, const float* __restrict__ post,
                    const unsigned short* __restrict__ Wb,   // [3][256*256] bf16
                    const float* __restrict__ bq, const float* __restrict__ bk,
                    const float* __restrict__ bv,
                    unsigned short* __restrict__ Qt,
                    unsigned short* __restrict__ Kt,
                    unsigned short* __restrict__ Vc) {
    const int mat   = blockIdx.z;
    const int b     = blockIdx.y;
    const int nBase = blockIdx.x * 64;

    const float* X = (mat == 0 ? pre : post) + (size_t)b * CHN * HWTOK;
    const unsigned short* W = Wb + (size_t)mat * CHN * CHN;
    const float* bias = (mat == 0 ? bq : (mat == 1 ? bk : bv));

    __shared__ unsigned short Xs[64][40];   // [token][c-chunk] bf16, padded rows

    const int tid   = threadIdx.x;
    const int wave  = tid >> 5;
    const int lane  = tid & 31;
    const int lmod  = lane & 15;
    const int g     = lane >> 4;
    const int rTile = wave >> 1;
    const int oHalf = wave & 1;

    v8f acc[8];
#pragma unroll
    for (int t = 0; t < 8; ++t) acc[t] = (v8f)0.0f;

    for (int kk = 0; kk < CHN; kk += 32) {
        __syncthreads();
        // stage X transposed: Xs[token][c] (f32 -> bf16), coalesced reads in n
#pragma unroll
        for (int i = 0; i < 8; ++i) {
            const int cc = (tid >> 6) + i * 4;   // 0..31
            const int nn = tid & 63;
            Xs[nn][cc] = f2bf(X[(size_t)(kk + cc) * HWTOK + nBase + nn]);
        }
        __syncthreads();

        if (mat < 2) {
            // M = token (A from LDS), N = out-channel (B from W rows)
            const v16bf xa = loadA_bf(&Xs[rTile * 16 + lmod][0], g);
#pragma unroll
            for (int t = 0; t < 8; ++t) {
                const int o = (oHalf * 8 + t) * 16 + lmod;
                const v16bf wb = loadB_bf(W + (size_t)o * CHN + kk + 16 * g);
                acc[t] = wmma_bf16(xa, wb, acc[t]);
            }
        } else {
            // M = out-channel (A from W rows), N = token (B from LDS)
            const v16bf xb = loadB_bf(&Xs[rTile * 16 + lmod][16 * g]);
#pragma unroll
            for (int t = 0; t < 8; ++t) {
                const int o = (oHalf * 8 + t) * 16 + lmod;
                const v16bf wa = loadA_bf(W + (size_t)o * CHN + kk, g);
                acc[t] = wmma_bf16(wa, xb, acc[t]);
            }
        }
    }

    if (mat < 2) {
        unsigned short* Out = (mat == 0 ? Qt : Kt) + (size_t)b * HWTOK * CHN;
#pragma unroll
        for (int t = 0; t < 8; ++t) {
            const int o  = (oHalf * 8 + t) * 16 + lmod;   // D lane = column = o
            const float bb = bias[o];
#pragma unroll
            for (int e = 0; e < 8; ++e) {
                const int token = nBase + rTile * 16 + e + 8 * g;  // D row
                Out[(size_t)token * CHN + o] = f2bf(acc[t][e] + bb);
            }
        }
    } else {
        unsigned short* Out = Vc + (size_t)b * CHN * HWTOK;
        const int token = nBase + rTile * 16 + lmod;      // D lane = column = n
#pragma unroll
        for (int t = 0; t < 8; ++t) {
#pragma unroll
            for (int e = 0; e < 8; ++e) {
                const int o = (oHalf * 8 + t) * 16 + e + 8 * g;    // D row
                Out[(size_t)o * HWTOK + token] = f2bf(acc[t][e] + bias[o]);
            }
        }
    }
}

// ---------------------------------------------------------------------------
// Phase 2: flash attention per wave (16 query rows), key tiles of 32.
//   S = Q K^T      (M=q, N=key, K=c; Q fragments preloaded, K streamed)
//   online softmax (row stats per (vgpr, half) via shfl_xor reductions)
//   O^T = V P^T    (M=c, N=q; P re-laid out through per-wave LDS)
//   out = gamma * O / l + pre_feat    (coalesced: D lanes = tokens)
// O rescale is a wave-uniform branch: taken only when the running max moved.
// ---------------------------------------------------------------------------
__global__ __launch_bounds__(256)
void flash_attn_wmma(const unsigned short* __restrict__ Qt,
                     const unsigned short* __restrict__ Kt,
                     const unsigned short* __restrict__ Vc,
                     const float* __restrict__ pre,
                     const float* __restrict__ gamma,
                     float* __restrict__ out) {
    const int b    = blockIdx.y;
    const int tid  = threadIdx.x;
    const int wave = tid >> 5;
    const int lane = tid & 31;
    const int lmod = lane & 15;
    const int g    = lane >> 4;
    const int qBase = (blockIdx.x * 8 + wave) * 16;

    const unsigned short* Q = Qt + (size_t)b * HWTOK * CHN;
    const unsigned short* K = Kt + (size_t)b * HWTOK * CHN;
    const unsigned short* V = Vc + (size_t)b * CHN * HWTOK;

    __shared__ unsigned short Ps[8][16 * 32];   // per-wave P tile (bf16)
    __shared__ float          St[8][16];        // per-wave row stats

    v8f O[16];
#pragma unroll
    for (int t = 0; t < 16; ++t) O[t] = (v8f)0.0f;
    float mrow[8], lrow[8];
#pragma unroll
    for (int v = 0; v < 8; ++v) { mrow[v] = -1.0e30f; lrow[v] = 0.0f; }

    // Q fragments are loop-invariant: preload all 8 (K = 0..255 in chunks of 32)
    const unsigned short* qrow = Q + (size_t)(qBase + lmod) * CHN;
    v16bf qa[8];
#pragma unroll
    for (int kk = 0; kk < 8; ++kk) qa[kk] = loadA_bf(qrow + kk * 32, g);

    for (int j = 0; j < HWTOK; j += 32) {
        // ---- S tiles: S0 = keys j..j+15, S1 = keys j+16..j+31 -------------
        v8f S0 = (v8f)0.0f, S1 = (v8f)0.0f;
        const unsigned short* krow0 = K + (size_t)(j + lmod) * CHN + 16 * g;
        const unsigned short* krow1 = K + (size_t)(j + 16 + lmod) * CHN + 16 * g;
#pragma unroll
        for (int kk = 0; kk < 8; ++kk) {
            const v16bf k0 = loadB_bf(krow0 + kk * 32);
            const v16bf k1 = loadB_bf(krow1 + kk * 32);
            S0 = wmma_bf16(qa[kk], k0, S0);
            S1 = wmma_bf16(qa[kk], k1, S1);
        }
        if (j + 32 < HWTOK)  // hint next K tile into cache (global_prefetch_b8)
            __builtin_prefetch(K + (size_t)(j + 32 + lmod) * CHN, 0, 1);

        // ---- online softmax (rows live at vgpr v, half g -> row v+8g) -----
#pragma unroll
        for (int v = 0; v < 8; ++v) {
            float rm = fmaxf(S0[v], S1[v]);
#pragma unroll
            for (int off = 1; off < 16; off <<= 1)
                rm = fmaxf(rm, __shfl_xor(rm, off, 32));
            const float mn  = fmaxf(mrow[v], rm);
            const float scv = __expf(mrow[v] - mn);
            mrow[v] = mn;
            const float p0 = __expf(S0[v] - mn);
            const float p1 = __expf(S1[v] - mn);
            float rs = p0 + p1;
#pragma unroll
            for (int off = 1; off < 16; off <<= 1)
                rs += __shfl_xor(rs, off, 32);
            lrow[v] = lrow[v] * scv + rs;
            // spill P (D layout: lane=key col, v=row) into LDS row-major
            Ps[wave][(v + 8 * g) * 32 + lmod]      = f2bf(p0);
            Ps[wave][(v + 8 * g) * 32 + 16 + lmod] = f2bf(p1);
            if (lmod == 0) St[wave][v + 8 * g] = scv;   // per-row rescale
        }
        asm volatile("s_wait_dscnt 0" ::: "memory");    // wave-local LDS RAW

        // Rescale O only when some row's max actually moved (wave-uniform).
        const float scl = St[wave][lmod];               // per token lane
        if (__any(scl != 1.0f)) {
#pragma unroll
            for (int t = 0; t < 16; ++t)
#pragma unroll
                for (int e = 0; e < 8; ++e) O[t][e] *= scl;
        }

        // P^T as B fragment: lane = token col, per-lane 16 contiguous j
        const v16bf pb = loadB_bf(&Ps[wave][lmod * 32 + 16 * g]);

        // ---- O^T accumulate: A = V rows (channel-major), B = P^T ----------
#pragma unroll
        for (int t = 0; t < 16; ++t) {
            const v16bf va =
                loadA_bf(V + (size_t)(t * 16 + lmod) * HWTOK + j, g);
            O[t] = wmma_bf16(va, pb, O[t]);
        }
    }

    // ---- epilogue: divide by l, gamma * O + residual ----------------------
    if (lmod == 0) {
#pragma unroll
        for (int v = 0; v < 8; ++v) St[wave][v + 8 * g] = lrow[v];
    }
    asm volatile("s_wait_dscnt 0" ::: "memory");
    const float linv = 1.0f / St[wave][lmod];
    const float gm   = gamma[0];
    const int token  = qBase + lmod;                    // D lane = token
    const float* preB = pre + (size_t)b * CHN * HWTOK;
    float*       outB = out + (size_t)b * CHN * HWTOK;
#pragma unroll
    for (int t = 0; t < 16; ++t) {
#pragma unroll
        for (int e = 0; e < 8; ++e) {
            const int c = t * 16 + e + 8 * g;           // D row = channel
            const size_t idx = (size_t)c * HWTOK + token;
            outB[idx] = gm * (O[t][e] * linv) + preB[idx];
        }
    }
}

// ---------------------------------------------------------------------------
extern "C" void kernel_launch(void* const* d_in, const int* in_sizes, int n_in,
                              void* d_out, int out_size, void* d_ws, size_t ws_size,
                              hipStream_t stream) {
    const float* pre   = (const float*)d_in[0];
    const float* post  = (const float*)d_in[1];
    const float* Wq    = (const float*)d_in[2];
    const float* bq    = (const float*)d_in[3];
    const float* Wk    = (const float*)d_in[4];
    const float* bk    = (const float*)d_in[5];
    const float* Wv    = (const float*)d_in[6];
    const float* bv    = (const float*)d_in[7];
    const float* gamma = (const float*)d_in[8];
    float* out = (float*)d_out;

    const size_t elems = (size_t)BATCH * HWTOK * CHN;   // 4.19M bf16 each
    unsigned short* Qt = (unsigned short*)d_ws;
    unsigned short* Kt = Qt + elems;
    unsigned short* Vc = Kt + elems;
    unsigned short* Wb = Vc + elems;                    // 3 * 256*256 bf16

    hipLaunchKernelGGL(prep_weights, dim3(3 * CHN * CHN / 256), dim3(256), 0,
                       stream, Wq, Wk, Wv, Wb);

    dim3 gridP(HWTOK / 64, BATCH, 3);   // (64, 4, 3)
    hipLaunchKernelGGL(proj_bf16_wmma, gridP, dim3(256), 0, stream,
                       pre, post, Wb, bq, bk, bv, Qt, Kt, Vc);

    dim3 gridA(HWTOK / 128, BATCH);     // (32, 4): 8 waves/block, 16 q-rows/wave
    hipLaunchKernelGGL(flash_attn_wmma, gridA, dim3(256), 0, stream,
                       Qt, Kt, Vc, pre, gamma, out);
}
